// DSS_3143916060956
// MI455X (gfx1250) — compile-verified, matrix operand order checked by reading, not run on hardware
//
#include <hip/hip_runtime.h>
#include <hip/hip_bf16.h>
#include <math.h>

// DSS layer for MI455X (gfx1250, wave32, WMMA + async global->LDS).
// H=512 hidden, N=64 states, B=4 batch, L=2048 sequence.
//
// Stage 1: cast lin_w fp32 -> bf16 (workspace)
// Stage 2: diagonal-SSM recurrence scan replaces the FFT convolution:
//          x_n[l] = exp(Lambda_n) * x_n[l-1] + u[l];  y[l] = Re(sum_n Wc_n x_n[l])
//          one wave per (b,h), 2 complex states per lane, intra-wave shfl reduce,
//          GELU, store bf16 activations as (B*L, H) row-major.
// Stage 3: WMMA bf16 GEMM (8192x512)@(512x512)^T + bias + GELU, store transposed
//          to out (B,H,L) fp32. B-operand slab staged in LDS via
//          global_load_async_to_lds_b128 (ASYNCcnt), padded rows to kill bank
//          conflicts; A-operand streamed from global (L2-resident).

typedef __attribute__((ext_vector_type(16))) __bf16 v16bf;
typedef __attribute__((ext_vector_type(8)))  __bf16 v8bf;
typedef __attribute__((ext_vector_type(8)))  float  v8f;

#define H_DIM   512
#define N_STATE 64
#define B_DIM   4
#define L_SEQ   2048
#define PADK    520   // 512 + 8 bf16 row padding: 1040 B stride = 4-bank skew

__device__ __forceinline__ float gelu_exact(float x) {
    return 0.5f * x * (1.0f + erff(x * 0.70710678118654752f));
}

// ---------------------------------------------------------------- Stage 1
__global__ void dss_cast_w_bf16(const float* __restrict__ w,
                                __bf16* __restrict__ wb, int n) {
    int i = blockIdx.x * blockDim.x + threadIdx.x;
    if (i < n) wb[i] = (__bf16)w[i];
}

// ---------------------------------------------------------------- Stage 2
// One wave32 per (b,h); lane owns states n=lane and n=lane+32.
__global__ __launch_bounds__(256) void dss_scan(
    const float* __restrict__ u,     // (B,H,L)
    const float* __restrict__ freq,  // (N,)
    const float* __restrict__ dec,   // (N,)
    const float* __restrict__ W,     // (H,N,2)
    __bf16* __restrict__ yT)         // (B*L, H) bf16 activations
{
    const int lane = threadIdx.x & 31;
    const int wave = threadIdx.x >> 5;
    const int bh   = blockIdx.x * 8 + wave;
    const int b    = bh / H_DIM;
    const int h    = bh % H_DIM;

    const int n0 = lane, n1 = lane + 32;
    // discrete multiplier exp(Lambda) = exp(-exp(decay)) * (cos f + i sin f)
    const float a0 = __expf(-__expf(dec[n0]));
    const float a1 = __expf(-__expf(dec[n1]));
    const float f0 = freq[n0], f1 = freq[n1];
    const float mr0 = a0 * __cosf(f0), mi0 = a0 * __sinf(f0);
    const float mr1 = a1 * __cosf(f1), mi1 = a1 * __sinf(f1);
    const float wr0 = W[(h * N_STATE + n0) * 2 + 0];
    const float wi0 = W[(h * N_STATE + n0) * 2 + 1];
    const float wr1 = W[(h * N_STATE + n1) * 2 + 0];
    const float wi1 = W[(h * N_STATE + n1) * 2 + 1];

    const float* up = u + (size_t)bh * L_SEQ;
    float xr0 = 0.f, xi0 = 0.f, xr1 = 0.f, xi1 = 0.f;

    for (int l = 0; l < L_SEQ; ++l) {
        const float uv = up[l];                       // broadcast load
        float nr0 = fmaf(mr0, xr0, fmaf(-mi0, xi0, uv));
        float ni0 = fmaf(mr0, xi0,  mi0 * xr0);
        float nr1 = fmaf(mr1, xr1, fmaf(-mi1, xi1, uv));
        float ni1 = fmaf(mr1, xi1,  mi1 * xr1);
        xr0 = nr0; xi0 = ni0; xr1 = nr1; xi1 = ni1;

        float c = fmaf(wr0, xr0, fmaf(-wi0, xi0, fmaf(wr1, xr1, -wi1 * xi1)));
        #pragma unroll
        for (int off = 16; off > 0; off >>= 1) c += __shfl_down(c, off, 32);
        if (lane == 0) {
            yT[((size_t)b * L_SEQ + l) * H_DIM + h] = (__bf16)gelu_exact(c);
        }
    }
}

// ---------------------------------------------------------------- Stage 3
// Z(M=8192, N=512) = gelu( A(M,K=512) @ Bw(N,K)^T + bias ), stored as (B,H,L).
// Block: 256 threads = 8 waves -> 128(M) x 64(N) tile; wave = 32x32 (2x2 WMMA).
// Whole 64-row B slab (64KB) staged once in LDS via async global->LDS.
__global__ __launch_bounds__(256) void dss_gemm_wmma(
    const __bf16* __restrict__ A,    // (8192, 512) row-major bf16
    const __bf16* __restrict__ Bw,   // lin_w (512, 512) row-major bf16
    const float*  __restrict__ bias, // (512,)
    float* __restrict__ out)         // (B, H, L) fp32
{
    constexpr int K = H_DIM;
    __shared__ __bf16 bsm[64 * PADK];                // 66,560 B of 320 KB WGP LDS

    const int lane = threadIdx.x & 31;
    const int wave = threadIdx.x >> 5;
    const int wm = wave & 3;                         // 4 waves along M
    const int wn = wave >> 2;                        // 2 waves along N
    const int m0 = blockIdx.x * 128 + wm * 32;
    const int n0 = blockIdx.y * 64  + wn * 32;
    const int nBlockBase = blockIdx.y * 64;

    // ---- async-stage B slab: 64 rows x 512 K bf16, 16B chunks, all 256 threads
    #pragma unroll
    for (int i = 0; i < 16; ++i) {
        const int c   = threadIdx.x + 256 * i;       // 0..4095 chunk id
        const int row = c >> 6;                      // 0..63
        const int kc  = c & 63;                      // 0..63 (x8 bf16 = 16 B)
        unsigned long long g =
            (unsigned long long)(const void*)(Bw + (size_t)(nBlockBase + row) * K + kc * 8);
        unsigned int l = (unsigned int)(__SIZE_TYPE__)&bsm[row * PADK + kc * 8];
        asm volatile("global_load_async_to_lds_b128 %0, %1, off"
                     :: "v"(l), "v"(g) : "memory");
    }
    asm volatile("s_wait_asynccnt 0" ::: "memory");  // ASYNCcnt drained
    __syncthreads();                                 // visible to all 8 waves

    const int half = lane >> 4;                      // ISA 16-bit A layout half
    const int mA0 = m0 + (lane & 15);
    const int mA1 = m0 + 16 + (lane & 15);
    const __bf16* brow0 = bsm + (size_t)(wn * 32 + (lane & 15)) * PADK;
    const __bf16* brow1 = brow0 + 16 * PADK;

    v8f c00 = {}, c01 = {}, c10 = {}, c11 = {};

    for (int k0 = 0; k0 < K; k0 += 32) {
        const int kb = k0 + half * 8;  // lane half 0: K{0..7,16..23}; half 1: K{8..15,24..31}
        v8bf a0lo = *(const v8bf*)(A + (size_t)mA0 * K + kb);
        v8bf a0hi = *(const v8bf*)(A + (size_t)mA0 * K + kb + 16);
        v8bf a1lo = *(const v8bf*)(A + (size_t)mA1 * K + kb);
        v8bf a1hi = *(const v8bf*)(A + (size_t)mA1 * K + kb + 16);
        v8bf b0lo = *(const v8bf*)(brow0 + kb);      // ds_load_b128, bank-skewed
        v8bf b0hi = *(const v8bf*)(brow0 + kb + 16);
        v8bf b1lo = *(const v8bf*)(brow1 + kb);
        v8bf b1hi = *(const v8bf*)(brow1 + kb + 16);

        v16bf a0 = __builtin_shufflevector(a0lo, a0hi, 0,1,2,3,4,5,6,7,8,9,10,11,12,13,14,15);
        v16bf a1 = __builtin_shufflevector(a1lo, a1hi, 0,1,2,3,4,5,6,7,8,9,10,11,12,13,14,15);
        v16bf b0 = __builtin_shufflevector(b0lo, b0hi, 0,1,2,3,4,5,6,7,8,9,10,11,12,13,14,15);
        v16bf b1 = __builtin_shufflevector(b1lo, b1hi, 0,1,2,3,4,5,6,7,8,9,10,11,12,13,14,15);

        c00 = __builtin_amdgcn_wmma_f32_16x16x32_bf16(false, a0, false, b0, (short)0, c00, false, false);
        c01 = __builtin_amdgcn_wmma_f32_16x16x32_bf16(false, a0, false, b1, (short)0, c01, false, false);
        c10 = __builtin_amdgcn_wmma_f32_16x16x32_bf16(false, a1, false, b0, (short)0, c10, false, false);
        c11 = __builtin_amdgcn_wmma_f32_16x16x32_bf16(false, a1, false, b1, (short)0, c11, false, false);
    }

    // C/D layout: VGPR r -> M = r + 8*half; N = lane%16. Fused bias+GELU+transpose.
    const int nB0 = n0 + (lane & 15);
    const int nB1 = n0 + 16 + (lane & 15);
    #define DSS_STORE_TILE(ACC, MBASE, NCOL)                                  \
        do {                                                                  \
            const float bval = bias[NCOL];                                    \
            _Pragma("unroll")                                                 \
            for (int r = 0; r < 8; ++r) {                                     \
                const int m = (MBASE) + r + half * 8;                         \
                const float v = gelu_exact(ACC[r] + bval);                    \
                const int bb = m >> 11;            /* L_SEQ = 2048 */         \
                const int l  = m & (L_SEQ - 1);                               \
                out[((size_t)bb * H_DIM + (NCOL)) * L_SEQ + l] = v;           \
            }                                                                 \
        } while (0)

    DSS_STORE_TILE(c00, m0,      nB0);
    DSS_STORE_TILE(c01, m0,      nB1);
    DSS_STORE_TILE(c10, m0 + 16, nB0);
    DSS_STORE_TILE(c11, m0 + 16, nB1);
    #undef DSS_STORE_TILE
}

// ---------------------------------------------------------------- launch
extern "C" void kernel_launch(void* const* d_in, const int* in_sizes, int n_in,
                              void* d_out, int out_size, void* d_ws, size_t ws_size,
                              hipStream_t stream) {
    const float* u     = (const float*)d_in[0];  // (B,H,L)
    const float* freq  = (const float*)d_in[1];  // (N,)
    const float* dec   = (const float*)d_in[2];  // (N,)
    const float* W     = (const float*)d_in[3];  // (H,N,2)
    const float* lin_w = (const float*)d_in[4];  // (H,H)
    const float* lin_b = (const float*)d_in[5];  // (H,)
    float* out = (float*)d_out;                  // (B,H,L)

    // Workspace layout: yT bf16 (B*L, H) = 8 MB, then lin_w bf16 = 512 KB.
    __bf16* yT = (__bf16*)d_ws;
    __bf16* wb = (__bf16*)((char*)d_ws +
                 (size_t)B_DIM * L_SEQ * H_DIM * sizeof(__bf16));

    dss_cast_w_bf16<<<(H_DIM * H_DIM + 255) / 256, 256, 0, stream>>>(
        lin_w, wb, H_DIM * H_DIM);

    dss_scan<<<(B_DIM * H_DIM) / 8, 256, 0, stream>>>(u, freq, dec, W, yT);

    dim3 grid(B_DIM * L_SEQ / 128, H_DIM / 64);
    dss_gemm_wmma<<<grid, 256, 0, stream>>>(yT, wb, lin_b, out);
}
